// MultiHeadAttention_3530463117423
// MI455X (gfx1250) — matrix operand order repaired and checked by reading, not compiled, not run on hardware
//
#include <hip/hip_runtime.h>

// ---------------- problem constants ----------------
constexpr int EMB = 2048;   // embed dim (= K and N of all GEMMs)
constexpr int NH  = 16;     // heads
constexpr int HD  = 128;    // head dim
constexpr int SEQ = 2048;   // sequence length
constexpr int BB  = 2;      // batch
constexpr int MT  = BB * SEQ; // 4096 rows for all GEMMs

typedef __attribute__((ext_vector_type(16))) __bf16 v16bf;
typedef __attribute__((ext_vector_type(8)))  __bf16 v8bf;
typedef __attribute__((ext_vector_type(8)))  float  v8f;

static __device__ __forceinline__ unsigned short f2bf(float f) {
    union { float f; unsigned int u; } v; v.f = f;
    unsigned int r = v.u + 0x7FFFu + ((v.u >> 16) & 1u);   // round-to-nearest-even
    return (unsigned short)(r >> 16);
}

// Assemble a 16-element bf16 fragment from two contiguous 8-element (16B) chunks.
static __device__ __forceinline__ v16bf frag16(const unsigned short* lo,
                                               const unsigned short* hi) {
    v8bf a = *(const v8bf*)lo;
    v8bf b = *(const v8bf*)hi;
    return __builtin_shufflevector(a, b, 0,1,2,3,4,5,6,7,8,9,10,11,12,13,14,15);
}

static __device__ __forceinline__ v8f zero8() {
    v8f z;
#pragma unroll
    for (int i = 0; i < 8; ++i) z[i] = 0.0f;
    return z;
}

// ---------------- fp32 -> bf16 conversion ----------------
__global__ void cvt_f32_bf16(const float* __restrict__ in,
                             unsigned short* __restrict__ out, int n) {
    int i = blockIdx.x * blockDim.x + threadIdx.x;
    int stride = gridDim.x * blockDim.x;
    for (; i < n; i += stride) out[i] = f2bf(in[i]);
}

// ---------------- bf16 WMMA GEMM: C[M,N] = A[M,K] * W[N,K]^T + bias ----------------
// Block tile 128(M) x 256(N), K-step 32, double-buffered LDS, one barrier/K-step.
// 8 waves: 2(M) x 4(N); each wave computes 64x64 = 4x4 wmma tiles (16 wmma/K-step).
__global__ __launch_bounds__(256) void gemm_bf16(
    const unsigned short* __restrict__ A,   // [4096,2048] bf16 row-major
    const unsigned short* __restrict__ W,   // [2048,2048] bf16 row-major (B = W^T)
    const float* __restrict__ bias,         // [2048]
    unsigned short* __restrict__ outB,      // bf16 strided output (mode 0)
    float* __restrict__ outF,               // f32 output (mode 1)
    int mode, long long sB, long long sH, long long sS, long long sD)
{
    __shared__ __align__(16) unsigned short As[2][128 * 40]; // pad 32->40 (bank spread)
    __shared__ __align__(16) unsigned short Bs[2][256 * 40];

    const int t     = threadIdx.x;
    const int lane  = t & 31;
    const int wid   = t >> 5;
    const int waveM = wid & 1;      // 2 waves along M (2*64 = 128)
    const int waveN = wid >> 1;     // 4 waves along N (4*64 = 256)
    const int m0 = blockIdx.x * 128;
    const int n0 = blockIdx.y * 256;
    const int hi = lane >> 4;       // half-wave selector for K chunks
    const int nl = lane & 15;

    v8f acc[4][4];
#pragma unroll
    for (int i = 0; i < 4; ++i)
#pragma unroll
        for (int j = 0; j < 4; ++j) acc[i][j] = zero8();

    const int lrow = t >> 2;        // 0..63
    const int lchk = (t & 3) * 8;   // 0,8,16,24 (16B chunks within 32-elem row)

    v8bf stA[2], stB[4];            // global->LDS staging registers

    // ---- prologue: load slab 0 into buffer 0 ----
    {
#pragma unroll
        for (int p = 0; p < 2; ++p)
            stA[p] = *(const v8bf*)&A[(size_t)(m0 + lrow + p * 64) * EMB + lchk];
#pragma unroll
        for (int p = 0; p < 4; ++p)
            stB[p] = *(const v8bf*)&W[(size_t)(n0 + lrow + p * 64) * EMB + lchk];
#pragma unroll
        for (int p = 0; p < 2; ++p)
            *(v8bf*)&As[0][(lrow + p * 64) * 40 + lchk] = stA[p];
#pragma unroll
        for (int p = 0; p < 4; ++p)
            *(v8bf*)&Bs[0][(lrow + p * 64) * 40 + lchk] = stB[p];
    }
    __syncthreads();

    const int NK = EMB / 32;
    for (int kt = 0; kt < NK; ++kt) {
        const int cur = kt & 1;
        const int k1 = (kt + 1) * 32;

        // issue next slab's global loads early (latency hidden under wmma burst)
        if (kt + 1 < NK) {
#pragma unroll
            for (int p = 0; p < 2; ++p)
                stA[p] = *(const v8bf*)&A[(size_t)(m0 + lrow + p * 64) * EMB + k1 + lchk];
#pragma unroll
            for (int p = 0; p < 4; ++p)
                stB[p] = *(const v8bf*)&W[(size_t)(n0 + lrow + p * 64) * EMB + k1 + lchk];
        }
        if (kt + 2 < NK) { // pull the slab after that toward the WGP caches
            __builtin_prefetch(&A[(size_t)(m0 + lrow) * EMB + k1 + 32 + lchk], 0, 1);
            __builtin_prefetch(&W[(size_t)(n0 + lrow) * EMB + k1 + 32 + lchk], 0, 1);
        }

        // A fragments: lanes<16 hold K 0-7 & 16-23, lanes>=16 hold K 8-15 & 24-31
        v16bf af[4];
#pragma unroll
        for (int mt = 0; mt < 4; ++mt) {
            const int r = waveM * 64 + mt * 16 + nl;
            af[mt] = frag16(&As[cur][r * 40 + hi * 8], &As[cur][r * 40 + hi * 8 + 16]);
        }
        // B fragments consumed one at a time: 1 LDS fragment load per 4 wmma
#pragma unroll
        for (int nt = 0; nt < 4; ++nt) {
            const int r = waveN * 64 + nt * 16 + nl;
            v16bf bfr = frag16(&Bs[cur][r * 40 + hi * 8], &Bs[cur][r * 40 + hi * 8 + 16]);
#pragma unroll
            for (int mt = 0; mt < 4; ++mt)
                acc[mt][nt] = __builtin_amdgcn_wmma_f32_16x16x32_bf16(
                    false, af[mt], false, bfr, (short)0, acc[mt][nt], false, false);
        }

        // stash next slab in the alternate buffer; single barrier per K-step
        if (kt + 1 < NK) {
#pragma unroll
            for (int p = 0; p < 2; ++p)
                *(v8bf*)&As[1 - cur][(lrow + p * 64) * 40 + lchk] = stA[p];
#pragma unroll
            for (int p = 0; p < 4; ++p)
                *(v8bf*)&Bs[1 - cur][(lrow + p * 64) * 40 + lchk] = stB[p];
        }
        __syncthreads();
    }

    // epilogue: C layout lane = N, VGPR r = row r (lanes<16) / r+8 (lanes>=16)
#pragma unroll
    for (int mt = 0; mt < 4; ++mt) {
#pragma unroll
        for (int nt = 0; nt < 4; ++nt) {
            const int n = n0 + waveN * 64 + nt * 16 + nl;
            const float bv = bias[n];
#pragma unroll
            for (int r = 0; r < 8; ++r) {
                const int m = m0 + waveM * 64 + mt * 16 + r + 8 * hi;
                const float val = acc[mt][nt][r] + bv;
                if (mode == 0) {
                    const long long bb = m >> 11, ss = m & 2047;
                    const long long hh = n >> 7,  dd = n & 127;
                    outB[bb * sB + hh * sH + ss * sS + dd * sD] = f2bf(val);
                } else {
                    outF[(size_t)m * EMB + n] = val;
                }
            }
        }
    }
}

// ---------------- flash attention: one wave per 16-row query tile ----------------
__global__ __launch_bounds__(256) void attn_flash(
    const unsigned short* __restrict__ Q,   // [B,H,S,D] bf16
    const unsigned short* __restrict__ Km,  // [B,H,S,D] bf16
    const unsigned short* __restrict__ Vt,  // [B,H,D,S] bf16 (transposed V)
    unsigned short* __restrict__ ctx)       // [B,S,E]   bf16
{
    __shared__ __align__(16) unsigned short Ps[8 * 16 * 40]; // per-wave P scratch
    const int t    = threadIdx.x;
    const int lane = t & 31;
    const int wid  = t >> 5;
    const int gw   = blockIdx.x * 8 + wid;   // 4096 waves total
    const int qt   = gw & 127;               // S/16 tiles
    const int h    = (gw >> 7) & 15;
    const int b    = gw >> 11;
    const int qs   = qt * 16;
    const int hi   = lane >> 4;
    const int nl   = lane & 15;
    unsigned short* pw = &Ps[wid * 16 * 40];

    const unsigned short* Qp = Q  + ((size_t)(b * NH + h) * SEQ) * HD;
    const unsigned short* Kp = Km + ((size_t)(b * NH + h) * SEQ) * HD;
    const unsigned short* Vp = Vt + ((size_t)(b * NH + h) * HD) * SEQ;

    // Q fragments: rows qs..qs+15, full head dim split into 4 K-chunks of 32
    v16bf qf[4];
#pragma unroll
    for (int j = 0; j < 4; ++j) {
        const unsigned short* base = Qp + (size_t)(qs + nl) * HD + j * 32 + hi * 8;
        qf[j] = frag16(base, base + 16);
    }

    v8f acc[8];
#pragma unroll
    for (int d = 0; d < 8; ++d) acc[d] = zero8();
    float mrow[8], lrw[8];
#pragma unroll
    for (int r = 0; r < 8; ++r) { mrow[r] = -1e30f; lrw[r] = 0.0f; }

    const float scale = 0.08838834764831845f; // 1/sqrt(128)
    const int nkb = (qs + 47) >> 5;           // key blocks covering keys 0..qs+15

    for (int kb = 0; kb < nkb; ++kb) {
        const int s0 = kb * 32;
        v8f c0 = zero8(), c1 = zero8();       // scores for keys s0..+15 and s0+16..+31
#pragma unroll
        for (int j = 0; j < 4; ++j) {
            const unsigned short* b0 = Kp + (size_t)(s0 + nl) * HD      + j * 32 + hi * 8;
            const unsigned short* b1 = Kp + (size_t)(s0 + 16 + nl) * HD + j * 32 + hi * 8;
            v16bf k0f = frag16(b0, b0 + 16);
            v16bf k1f = frag16(b1, b1 + 16);
            c0 = __builtin_amdgcn_wmma_f32_16x16x32_bf16(false, qf[j], false, k0f, (short)0, c0, false, false);
            c1 = __builtin_amdgcn_wmma_f32_16x16x32_bf16(false, qf[j], false, k1f, (short)0, c1, false, false);
        }

        const bool needmask = (s0 + 31 > qs);  // uniform per wave
        const int key0 = s0 + nl, key1 = s0 + 16 + nl;
        float fsc[8];
#pragma unroll
        for (int r = 0; r < 8; ++r) {
            const int rowm = qs + r + 8 * hi;
            float s0v = c0[r] * scale;
            float s1v = c1[r] * scale;
            if (needmask) {
                s0v = (key0 <= rowm) ? s0v : -1e30f;
                s1v = (key1 <= rowm) ? s1v : -1e30f;
            }
            float bm = fmaxf(s0v, s1v);
#pragma unroll
            for (int off = 1; off < 16; off <<= 1)
                bm = fmaxf(bm, __shfl_xor(bm, off, 16));
            const float mnew = fmaxf(mrow[r], bm);
            const float f    = __expf(mrow[r] - mnew);
            const float p0   = __expf(s0v - mnew);
            const float p1   = __expf(s1v - mnew);
            float rs = p0 + p1;
#pragma unroll
            for (int off = 1; off < 16; off <<= 1)
                rs += __shfl_xor(rs, off, 16);
            lrw[r]  = lrw[r] * f + rs;
            mrow[r] = mnew;
            fsc[r]  = f;
            // C-layout -> LDS (row-major P tile, 16x32, padded stride 40)
            pw[(r + 8 * hi) * 40 + nl]      = f2bf(p0);
            pw[(r + 8 * hi) * 40 + nl + 16] = f2bf(p1);
        }
#pragma unroll
        for (int d = 0; d < 8; ++d)
#pragma unroll
            for (int r = 0; r < 8; ++r)
                acc[d][r] *= fsc[r];

        // P as A-fragment (same wave wrote LDS; DS ops are in-order within a wave)
        v16bf pf = frag16(&pw[nl * 40 + hi * 8], &pw[nl * 40 + hi * 8 + 16]);
#pragma unroll
        for (int d = 0; d < 8; ++d) {
            const unsigned short* vb = Vp + (size_t)(d * 16 + nl) * SEQ + s0 + hi * 8;
            v16bf vf = frag16(vb, vb + 16);   // V^T is K-contiguous per output dim
            acc[d] = __builtin_amdgcn_wmma_f32_16x16x32_bf16(
                false, pf, false, vf, (short)0, acc[d], false, false);
        }
    }

    float inv[8];
#pragma unroll
    for (int r = 0; r < 8; ++r) inv[r] = 1.0f / lrw[r];
#pragma unroll
    for (int d = 0; d < 8; ++d) {
#pragma unroll
        for (int r = 0; r < 8; ++r) {
            const int m = qs + r + 8 * hi;
            const float val = acc[d][r] * inv[r];
            ctx[((size_t)(b * SEQ + m)) * EMB + h * HD + d * 16 + nl] = f2bf(val);
        }
    }
}

// ---------------- host-side orchestration ----------------
extern "C" void kernel_launch(void* const* d_in, const int* in_sizes, int n_in,
                              void* d_out, int out_size, void* d_ws, size_t ws_size,
                              hipStream_t stream) {
    (void)in_sizes; (void)n_in; (void)out_size; (void)ws_size;
    const float* x  = (const float*)d_in[0];
    const float* Wq = (const float*)d_in[1];
    const float* bq = (const float*)d_in[2];
    const float* Wk = (const float*)d_in[3];
    const float* bk = (const float*)d_in[4];
    const float* Wv = (const float*)d_in[5];
    const float* bv = (const float*)d_in[6];
    const float* Wo = (const float*)d_in[7];
    const float* bo = (const float*)d_in[8];

    unsigned short* ws  = (unsigned short*)d_ws;
    const size_t NX = (size_t)MT * EMB;      // 8,388,608
    const size_t NW = (size_t)EMB * EMB;     // 4,194,304
    unsigned short* xb  = ws;
    unsigned short* wqb = xb  + NX;
    unsigned short* wkb = wqb + NW;
    unsigned short* wvb = wkb + NW;
    unsigned short* wob = wvb + NW;
    unsigned short* Qb  = wob + NW;
    unsigned short* Kb  = Qb  + NX;
    unsigned short* Vtb = Kb  + NX;
    unsigned short* ctx = Vtb + NX;

    cvt_f32_bf16<<<2048, 256, 0, stream>>>(x,  xb,  (int)NX);
    cvt_f32_bf16<<<1024, 256, 0, stream>>>(Wq, wqb, (int)NW);
    cvt_f32_bf16<<<1024, 256, 0, stream>>>(Wk, wkb, (int)NW);
    cvt_f32_bf16<<<1024, 256, 0, stream>>>(Wv, wvb, (int)NW);
    cvt_f32_bf16<<<1024, 256, 0, stream>>>(Wo, wob, (int)NW);

    dim3 g(MT / 128, EMB / 256);  // 32 x 8 blocks
    // Q,K -> [B,H,S,D] ; strides in elements
    const long long sB = (long long)NH * SEQ * HD, sH = (long long)SEQ * HD;
    gemm_bf16<<<g, 256, 0, stream>>>(xb, wqb, bq, Qb,  nullptr, 0, sB, sH, (long long)HD, 1LL);
    gemm_bf16<<<g, 256, 0, stream>>>(xb, wkb, bk, Kb,  nullptr, 0, sB, sH, (long long)HD, 1LL);
    // V -> [B,H,D,S] (transposed for the PV B-fragment)
    gemm_bf16<<<g, 256, 0, stream>>>(xb, wvb, bv, Vtb, nullptr, 0, sB, sH, 1LL, (long long)SEQ);

    attn_flash<<<512, 256, 0, stream>>>(Qb, Kb, Vtb, ctx);

    // out = ctx @ Wo^T + bo (fp32)
    gemm_bf16<<<g, 256, 0, stream>>>(ctx, wob, bo, nullptr, (float*)d_out, 1, 0, 0, 0, 0);
}
